// RNN_86938728006113
// MI455X (gfx1250) — matrix operand order repaired and checked by reading, not compiled, not run on hardware
//
#include <hip/hip_runtime.h>
#include <hip/hip_bf16.h>
#include <stdint.h>

// LSTM: N=64, T=512, d=512, H=1024, O=1
//  1) cvt_kernel:    x, w_ih, w_hh fp32 -> bf16 (HBM-bound); zero barrier ctrs.
//  2) xproj_kernel:  x_proj = x @ w_ih^T + b  (WMMA bf16 -> f32), ping-pong
//                    double-buffered fragment loads, bf16 [t][n][4H].
//  3) lstm_kernel:   persistent 32 blocks; per step WMMA gates = h @ w_hh_slice^T,
//                    h staged into LDS via TDM tensor_load_to_lds, c lives in LDS
//                    across all 512 steps, fused cell update, grid sense barrier.
//  4) fc_kernel:     y = sigmoid(h_T @ fc_w^T + fc_b).

typedef __attribute__((ext_vector_type(16))) __bf16 v16bf;
typedef __attribute__((ext_vector_type(8)))  float  v8f;
typedef __attribute__((ext_vector_type(4)))  unsigned int u32x4;
typedef __attribute__((ext_vector_type(8)))  int i32x8;
typedef __attribute__((ext_vector_type(4)))  int i32x4;

#define N_B 64
#define T_S 512
#define D_I 512
#define H_D 1024
#define G4  4096

#define NB_REC 32            // persistent blocks
#define HS     (H_D/NB_REC)  // 32 H-columns per block

#if defined(__has_builtin)
#if __has_builtin(__builtin_amdgcn_tensor_load_to_lds) && __has_builtin(__builtin_amdgcn_s_wait_tensorcnt)
#define HAVE_TDM 1
#endif
#endif

__device__ __forceinline__ unsigned short f2bf(float f){
  union { float f; unsigned int u; } x; x.f = f;
  unsigned int u = x.u;
  unsigned int r = u + 0x7FFFu + ((u >> 16) & 1u);   // round-to-nearest-even
  return (unsigned short)(r >> 16);
}
__device__ __forceinline__ float bf2f(unsigned short h){
  union { unsigned int u; float f; } x; x.u = ((unsigned int)h) << 16;
  return x.f;
}
__device__ __forceinline__ unsigned int packbf(float lo, float hi){
  return (unsigned int)f2bf(lo) | ((unsigned int)f2bf(hi) << 16);
}
__device__ __forceinline__ float sigmoidf_(float x){ return 1.0f / (1.0f + __expf(-x)); }

union Frag16 { unsigned int u[8]; uint4 q[2]; v16bf v; };

// A fragment (16-bit 16x32, ISA 7.12.2): per lane row fixed, K = k0+8*half+{0..7}
// in V0..3 and k0+16+8*half+{0..7} in V4..7 -> b128 at +8*half and +16+8*half.
__device__ __forceinline__ Frag16 load_a(const unsigned short* row, int k0, int kh){
  Frag16 a;
  const unsigned short* p = row + k0 + 8*kh;
  a.q[0] = *(const uint4*)(p);
  a.q[1] = *(const uint4*)(p + 16);
  return a;
}
// B fragment (16-bit 32x16): per lane column fixed, K = k0+8*half+{0..7} in V0..3
// and k0+16+8*half+{0..7} in V4..7 (same as A) -> b128 at +8*half and +16+8*half.
__device__ __forceinline__ Frag16 load_b(const unsigned short* col, int k0, int kh){
  Frag16 b;
  const unsigned short* p = col + k0 + 8*kh;
  b.q[0] = *(const uint4*)(p);
  b.q[1] = *(const uint4*)(p + 16);
  return b;
}

// ---------------------------------------------------------------- cvt -------
__global__ void cvt_kernel(const float* __restrict__ x,
                           const float* __restrict__ wih, const float* __restrict__ whh,
                           unsigned int* __restrict__ xb2,
                           unsigned int* __restrict__ wihb2, unsigned int* __restrict__ whhb2,
                           unsigned int* __restrict__ gsync)
{
  size_t tid    = (size_t)blockIdx.x * blockDim.x + threadIdx.x;
  size_t stride = (size_t)gridDim.x * blockDim.x;
  const size_t n_x   = ((size_t)N_B * T_S * D_I) / 2;  // pairs
  const size_t n_whh = ((size_t)G4 * H_D) / 2;
  const size_t n_wih = ((size_t)G4 * D_I) / 2;
  const float2* x2 = (const float2*)x;
  const float2* wih2 = (const float2*)wih;
  const float2* whh2 = (const float2*)whh;
  for (size_t i = tid; i < n_x; i += stride){
    float2 v = x2[i];
    xb2[i] = packbf(v.x, v.y);
    if (i < n_whh){ float2 w = whh2[i]; whhb2[i] = packbf(w.x, w.y); }
    if (i < n_wih){ float2 w = wih2[i]; wihb2[i] = packbf(w.x, w.y); }
  }
  if (tid == 0){ gsync[0] = 0u; gsync[1] = 0u; }
}

// ------------------------------------------------------------- x-proj -------
// Block = 4 waves; wave w -> row-tile (blockIdx.x*4 + w), 4 col-tiles (A reuse).
// Ping-pong double buffering: fragment set for k-step kb+1 loads while kb's
// WMMAs execute; no rotation copies.
__global__ __launch_bounds__(128) void xproj_kernel(
    const unsigned short* __restrict__ xb, const unsigned short* __restrict__ wihb,
    const float* __restrict__ b_ih, const float* __restrict__ b_hh,
    unsigned short* __restrict__ xproj)
{
  const int wave = threadIdx.x >> 5, lane = threadIdx.x & 31;
  const int lm = lane & 15, kh = lane >> 4;
  const int r0 = blockIdx.x * 64 + wave * 16;   // flattened (n*T + t) row base
  const int g0 = blockIdx.y * 64;               // gate-column base

  const v8f vz = {0.f,0.f,0.f,0.f,0.f,0.f,0.f,0.f};
  v8f acc[4];
  #pragma unroll
  for (int c = 0; c < 4; ++c) acc[c] = vz;

  const unsigned short* xrow = xb + (size_t)(r0 + lm) * D_I;
  const unsigned short* wcol[4];
  #pragma unroll
  for (int c = 0; c < 4; ++c) wcol[c] = wihb + (size_t)(g0 + c*16 + lm) * D_I;

  Frag16 aP, bP[4], aQ, bQ[4];                  // ping (P) / pong (Q) sets
  aP = load_a(xrow, 0, kh);
  #pragma unroll
  for (int c = 0; c < 4; ++c) bP[c] = load_b(wcol[c], 0, kh);

  #pragma unroll
  for (int kb = 0; kb < D_I/32; kb += 2){
    if (kb + 1 < D_I/32){                       // prefetch kb+1 into Q
      const int kn = (kb + 1) * 32;
      aQ = load_a(xrow, kn, kh);
      #pragma unroll
      for (int c = 0; c < 4; ++c) bQ[c] = load_b(wcol[c], kn, kh);
    }
    #pragma unroll
    for (int c = 0; c < 4; ++c)
      acc[c] = __builtin_amdgcn_wmma_f32_16x16x32_bf16(
                 false, aP.v, false, bP[c].v, (short)0, acc[c], false, false);
    if (kb + 2 < D_I/32){                       // prefetch kb+2 into P
      const int kn = (kb + 2) * 32;
      aP = load_a(xrow, kn, kh);
      #pragma unroll
      for (int c = 0; c < 4; ++c) bP[c] = load_b(wcol[c], kn, kh);
    }
    if (kb + 1 < D_I/32){
      #pragma unroll
      for (int c = 0; c < 4; ++c)
        acc[c] = __builtin_amdgcn_wmma_f32_16x16x32_bf16(
                   false, aQ.v, false, bQ[c].v, (short)0, acc[c], false, false);
    }
  }

  // store: row m -> (n = m>>9, t = m & 511); layout [t][n][4H] bf16
  #pragma unroll
  for (int c = 0; c < 4; ++c){
    const int g = g0 + c*16 + lm;
    const float bias = b_ih[g] + b_hh[g];
    #pragma unroll
    for (int e = 0; e < 8; ++e){
      const int m = r0 + e + 8*kh;
      const int t = m & (T_S - 1);
      const int n = m >> 9;
      xproj[((size_t)t * N_B + n) * G4 + g] = f2bf(acc[c][e] + bias);
    }
  }
}

// ------------------------------------------------------------ recurrence ----
__global__ __launch_bounds__(128) void lstm_kernel(
    const unsigned short* __restrict__ whhb,
    const unsigned short* __restrict__ xproj,
    unsigned short* __restrict__ hbuf,       // 2 * N_B*H_D bf16 (double buffer)
    unsigned int* __restrict__ gsync)
{
  __shared__ unsigned short h_lds[N_B * H_D]; // 128 KB (full h, staged per step)
  __shared__ float          c_lds[N_B * HS];  //   8 KB (block's c slice, persistent)

  const int wave = threadIdx.x >> 5, lane = threadIdx.x & 31;
  const int lm = lane & 15, kh = lane >> 4;
  const int h0 = blockIdx.x * HS;

  for (int i = threadIdx.x; i < N_B*HS; i += 128) c_lds[i] = 0.0f;
  {
    uint4 z; z.x = z.y = z.z = z.w = 0u;
    uint4* hq = (uint4*)h_lds;
    for (int i = threadIdx.x; i < (N_B*H_D)/8; i += 128) hq[i] = z;  // h_0 = 0
  }
  __syncthreads();

#if HAVE_TDM
  // Tensor DMA descriptors: 2-D tile (1024 x 64) of 2-byte elements, contiguous.
  // group0: [1:0]=count(1) | [63:32]=lds_addr | [120:64]=global_addr | [127:126]=type(2)
  const unsigned int lds_base = (unsigned int)(uintptr_t)(void*)h_lds;
  u32x4 g0buf[2];
  #pragma unroll
  for (int b2 = 0; b2 < 2; ++b2){
    unsigned long long a =
        (unsigned long long)(uintptr_t)(hbuf + (size_t)b2 * (N_B * H_D));
    u32x4 g0; g0[0] = 1u; g0[1] = lds_base;
    g0[2] = (unsigned int)(a & 0xFFFFFFFFu);
    g0[3] = (unsigned int)((a >> 32) & 0x01FFFFFFu) | (2u << 30);
    g0buf[b2] = g0;
  }
  // group1: data_size=1(2B); tensor_dim0=1024; tensor_dim1=64; tile_dim0=1024;
  //         tile_dim1=64; tensor_dim0_stride=1024
  i32x8 g1d;
  g1d[0] = (int)(1u << 16);
  g1d[1] = (int)((H_D & 0xFFFF) << 16);
  g1d[2] = (int)((H_D >> 16) | ((N_B & 0xFFFF) << 16));
  g1d[3] = (int)((N_B >> 16) | (H_D << 16));
  g1d[4] = (int)N_B;
  g1d[5] = (int)H_D;
  g1d[6] = 0; g1d[7] = 0;
  i32x4 gz4; gz4[0] = gz4[1] = gz4[2] = gz4[3] = 0;
  i32x8 gz8; gz8[0]=gz8[1]=gz8[2]=gz8[3]=gz8[4]=gz8[5]=gz8[6]=gz8[7]=0;
#endif

  const int m0 = wave * 16;                         // wave's 16 batch rows
  const unsigned short* arow_p = h_lds + (size_t)(m0 + lm) * H_D;
  const unsigned short* wcol[8];
  #pragma unroll
  for (int tc = 0; tc < 8; ++tc){
    const int q = tc >> 1, j = tc & 1;
    wcol[tc] = whhb + (size_t)(q*H_D + h0 + j*16 + lm) * H_D;
  }

  for (int t = 0; t < T_S; ++t){
    const v8f vz = {0.f,0.f,0.f,0.f,0.f,0.f,0.f,0.f};
    v8f acc[8];                                     // 4 gates x 2 col-tiles
    #pragma unroll
    for (int i = 0; i < 8; ++i) acc[i] = vz;

    for (int kb = 0; kb < H_D/32; ++kb){
      const int k0 = kb*32;
      Frag16 a = load_a(arow_p, k0, kh);            // from LDS (ds_load_b128 x2)
      if ((kb & 7) == 0 && kb + 8 < H_D/32)         // keep w_hh hot in near cache
        __builtin_prefetch((const void*)(wcol[0] + k0 + 256), 0, 1);
      #pragma unroll
      for (int tc = 0; tc < 8; ++tc){
        Frag16 b = load_b(wcol[tc], k0, kh);
        acc[tc] = __builtin_amdgcn_wmma_f32_16x16x32_bf16(
                    false, a.v, false, b.v, (short)0, acc[tc], false, false);
      }
    }

    // fused LSTM cell: i/f/g/o for (row,col) all live in this lane/element
    unsigned short* hout = hbuf + (size_t)((t + 1) & 1) * (N_B * H_D);
    const unsigned short* xp = xproj + (size_t)t * N_B * G4;
    #pragma unroll
    for (int j = 0; j < 2; ++j){
      const int cl = j*16 + lm;
      #pragma unroll
      for (int e = 0; e < 8; ++e){
        const int m = m0 + e + 8*kh;
        const size_t xbo = (size_t)m * G4 + h0 + cl;
        float iv = sigmoidf_(acc[0 + j][e] + bf2f(xp[xbo          ]));
        float fv = sigmoidf_(acc[2 + j][e] + bf2f(xp[xbo +   H_D  ]));
        float gv = tanhf    (acc[4 + j][e] + bf2f(xp[xbo + 2*H_D  ]));
        float ov = sigmoidf_(acc[6 + j][e] + bf2f(xp[xbo + 3*H_D  ]));
        float cn = fv * c_lds[m*HS + cl] + iv * gv;
        c_lds[m*HS + cl] = cn;
        hout[(size_t)m * H_D + h0 + cl] = f2bf(ov * tanhf(cn));
      }
    }
    __threadfence();

    // ------- grid-wide sense barrier (persistent kernel sync) -------
    __syncthreads();
    if (threadIdx.x == 0){
      volatile unsigned int* gen = gsync + 1;
      const unsigned int g = *gen;
      const unsigned int arr = atomicAdd(gsync, 1u) + 1u;
      if (arr == NB_REC){
        atomicExch(gsync, 0u);
        __threadfence();
        atomicAdd((unsigned int*)(gsync + 1), 1u);
      } else {
        while (*gen == g) __builtin_amdgcn_s_sleep(2);
      }
    }
    __syncthreads();

    // stage next-step h into LDS (written by all blocks, read through L2)
    if (t + 1 < T_S){
#if HAVE_TDM
      if (threadIdx.x < 32){                        // wave 0 drives the DMA
        __builtin_amdgcn_tensor_load_to_lds(g0buf[(t + 1) & 1], g1d, gz4, gz4, gz8, 0);
        __builtin_amdgcn_s_wait_tensorcnt(0);
      }
      __syncthreads();
#else
      const uint4* hin = (const uint4*)(hbuf + (size_t)((t + 1) & 1) * (N_B * H_D));
      uint4* hl = (uint4*)h_lds;
      for (int i = threadIdx.x; i < (N_B*H_D)/8; i += 128) hl[i] = hin[i];
      __syncthreads();
#endif
    }
  }
}

// ----------------------------------------------------------------- FC -------
__global__ __launch_bounds__(128) void fc_kernel(
    const unsigned short* __restrict__ hT, const float* __restrict__ fc_w,
    const float* __restrict__ fc_b, float* __restrict__ out)
{
  const int n = blockIdx.x;
  __shared__ float red[128];
  float s = 0.f;
  for (int k = threadIdx.x; k < H_D; k += 128)
    s += bf2f(hT[(size_t)n * H_D + k]) * fc_w[k];
  red[threadIdx.x] = s;
  __syncthreads();
  for (int off = 64; off > 0; off >>= 1){
    if (threadIdx.x < off) red[threadIdx.x] += red[threadIdx.x + off];
    __syncthreads();
  }
  if (threadIdx.x == 0) out[n] = 1.f / (1.f + __expf(-(red[0] + fc_b[0])));
}

// ------------------------------------------------------------- launch -------
extern "C" void kernel_launch(void* const* d_in, const int* in_sizes, int n_in,
                              void* d_out, int out_size, void* d_ws, size_t ws_size,
                              hipStream_t stream)
{
  (void)in_sizes; (void)n_in; (void)out_size; (void)ws_size;
  const float* x    = (const float*)d_in[0];
  const float* w_ih = (const float*)d_in[1];
  const float* w_hh = (const float*)d_in[2];
  const float* b_ih = (const float*)d_in[3];
  const float* b_hh = (const float*)d_in[4];
  const float* fc_w = (const float*)d_in[5];
  const float* fc_b = (const float*)d_in[6];
  float* out = (float*)d_out;

  // ws layout (bytes, all 256-aligned):
  //   wihb  bf16 [4096 x  512] :           0 ..   4,194,304
  //   whhb  bf16 [4096 x 1024] :   4,194,304 ..  12,582,912
  //   xb    bf16 [64x512x512]  :  12,582,912 ..  46,137,344
  //   xproj bf16 [512x64x4096] :  46,137,344 .. 314,572,800
  //   hbuf  bf16 2x[64 x 1024] : 314,572,800 .. 314,834,944
  //   gsync u32 [2]            : 314,834,944 ..
  char* ws = (char*)d_ws;
  unsigned short* wihb  = (unsigned short*)(ws);
  unsigned short* whhb  = (unsigned short*)(ws + 4194304);
  unsigned short* xb    = (unsigned short*)(ws + 12582912);
  unsigned short* xproj = (unsigned short*)(ws + 46137344ull);
  unsigned short* hbuf  = (unsigned short*)(ws + 314572800ull);
  unsigned int*   gsync = (unsigned int*)  (ws + 314834944ull);

  cvt_kernel<<<2048, 256, 0, stream>>>(x, w_ih, w_hh,
                                       (unsigned int*)xb, (unsigned int*)wihb,
                                       (unsigned int*)whhb, gsync);

  dim3 gproj(512, 64);                           // (32768/64) x (4096/64)
  xproj_kernel<<<gproj, 128, 0, stream>>>(xb, wihb, b_ih, b_hh, xproj);

  lstm_kernel<<<NB_REC, 128, 0, stream>>>(whhb, xproj, hbuf, gsync);

  // T=512 even -> final h is in buffer 0
  fc_kernel<<<64, 128, 0, stream>>>(hbuf, fc_w, fc_b, out);
}
